// QuantumHDNet_14602888806428
// MI455X (gfx1250) — matrix operand order repaired and checked by reading, not compiled, not run on hardware
//
#include <hip/hip_runtime.h>
#include <hip/hip_bf16.h>
#include <math.h>

// ---------------------------------------------------------------------------
// QuantumHDNet forward on MI455X (gfx1250, wave32, WMMA).
//  * GEMMs: v_wmma_f32_16x16x32_f16, one wave owns a 16x128 (or 16x64) output
//    row-stripe -> A fragment loaded once, 8-16 WMMAs issued per K-step.
//  * h kept in f32 (residual/LN precision) + f16 shadow (WMMA A operand,
//    loaded as two aligned b128 per lane, zero cvt).
//  * Edge aggregation: wave-per-edge float4 gather + global_atomic_add_f32.
//  * inv_deg folded into the agg A-operand f32->f16 conversion.
// ---------------------------------------------------------------------------

typedef __attribute__((ext_vector_type(16))) _Float16 v16h;
typedef __attribute__((ext_vector_type(8)))  _Float16 v8h;
typedef __attribute__((ext_vector_type(8)))  float    v8f;

#define N_NODES 50000
#define N_EDGES 800000
#define DHID    128
#define DIN     64
#define SUNDIM  80
#define FILMH   64
#define NLAYERS 3
#define MTILES  3125  // N_NODES / 16
#define PB      128   // partial-reduction blocks for attention pools

#define WMMA_F16(a, b, c) \
  __builtin_amdgcn_wmma_f32_16x16x32_f16(false, (a), false, (b), (short)0, (c), false, false)

// ---------------- wave32 reduction ----------------
static __device__ __forceinline__ float wave_sum(float v) {
#pragma unroll
  for (int m = 16; m >= 1; m >>= 1) v += __shfl_xor(v, m, 32);
  return v;
}

// ---------------- WMMA fragment helpers ----------------
// 16-bit A 16x32 ISA layout: lane<16 -> M=lane, K in {0..7,16..23};
// lane>=16 -> same M, K in {8..15,24..31}.  v16h element j: two contiguous
// 8-half chunks at k0 + 8*half and k0 + 16 + 8*half.
static __device__ __forceinline__ v16h load_a_f32(const float* __restrict__ A,
                                                  int lda, int m0, int k0,
                                                  float scale) {
  const int lane = threadIdx.x & 31;
  const int half = lane >> 4;
  const float* row = A + (size_t)(m0 + (lane & 15)) * lda + k0 + (half << 3);
  v16h a;
#pragma unroll
  for (int j = 0; j < 8; ++j) {
    a[j]     = (_Float16)(row[j] * scale);
    a[j + 8] = (_Float16)(row[j + 16] * scale);
  }
  return a;
}

// f16 A operand: two aligned 16B loads per lane, no conversion.
static __device__ __forceinline__ v16h load_a_f16(const _Float16* __restrict__ A,
                                                  int lda, int m0, int k0) {
  const int lane = threadIdx.x & 31;
  const int half = lane >> 4;
  const _Float16* row = A + (size_t)(m0 + (lane & 15)) * lda + k0 + (half << 3);
  const v8h lo = *(const v8h*)(row);
  const v8h hi = *(const v8h*)(row + 16);
  v16h a;
#pragma unroll
  for (int j = 0; j < 8; ++j) { a[j] = lo[j]; a[j + 8] = hi[j]; }
  return a;
}

// B (32x16 f16) pre-packed: each lane's fragment is one contiguous v16h.
static __device__ __forceinline__ v16h load_b_pack(const _Float16* __restrict__ Bp,
                                                   int ntiles, int kt, int nt) {
  const int lane = threadIdx.x & 31;
  return *(const v16h*)(Bp + (((size_t)(kt * ntiles + nt) * 32 + lane) << 4));
}

// C/D 16x16 f32: VGPR r -> row m0 + r + half*8, col n0 + (lane&15).
static __device__ __forceinline__ void store_tile(float* __restrict__ C,
                                                  _Float16* __restrict__ Ch,
                                                  int ldc, int m0, int n0, v8f c,
                                                  const float* __restrict__ bias,
                                                  int relu) {
  const int lane = threadIdx.x & 31;
  const int half = lane >> 4;
  const int col = n0 + (lane & 15);
  const float b = bias ? bias[col] : 0.0f;
#pragma unroll
  for (int r = 0; r < 8; ++r) {
    float v = c[r] + b;
    if (relu) v = fmaxf(v, 0.0f);
    const size_t idx = (size_t)(m0 + r + half * 8) * ldc + col;
    C[idx] = v;
    if (Ch) Ch[idx] = (_Float16)v;
  }
}

// ---------------- weight pre-pack (f32 [K,N] -> fragment-ordered f16) -------
__global__ void k_pack_weight(const float* __restrict__ W, int K, int Ncols,
                              _Float16* __restrict__ out) {
  const int idx = blockIdx.x * blockDim.x + threadIdx.x;
  const int total = (K >> 5) * (Ncols >> 4) * 512;
  if (idx >= total) return;
  const int j = idx & 15;
  const int lane = (idx >> 4) & 31;
  const int tile = idx >> 9;
  const int ntiles = Ncols >> 4;
  const int nt = tile % ntiles;
  const int kt = tile / ntiles;
  const int k = (kt << 5) + ((j >> 3) << 4) + ((lane >> 4) << 3) + (j & 7);
  const int n = (nt << 4) + (lane & 15);
  out[idx] = (_Float16)W[(size_t)k * Ncols + n];
}

// ---- input GEMM: h = relu(x @ W_in + b_in); wave owns 16x128 stripe --------
__global__ void k_gemm_in(const float* __restrict__ x,
                          const _Float16* __restrict__ Bp,
                          const float* __restrict__ bias,
                          float* __restrict__ h, _Float16* __restrict__ hh) {
  const int mt = (blockIdx.x * blockDim.x + threadIdx.x) >> 5;
  if (mt >= MTILES) return;
  v8f c[8] = {};
#pragma unroll
  for (int kt = 0; kt < (DIN >> 5); ++kt) {
    const v16h a = load_a_f32(x, DIN, mt << 4, kt << 5, 1.0f);
#pragma unroll
    for (int nt = 0; nt < 8; ++nt)
      c[nt] = WMMA_F16(a, load_b_pack(Bp, 8, kt, nt), c[nt]);
  }
#pragma unroll
  for (int nt = 0; nt < 8; ++nt)
    store_tile(h, hh, DHID, mt << 4, nt << 4, c[nt], bias, 1);
}

// ---- layer GEMM: hnew = (agg*invdeg)@Wl + h@Wr + bl; 64 WMMAs per wave ----
__global__ void k_gemm_layer(const float* __restrict__ agg,
                             const float* __restrict__ invdeg,
                             const _Float16* __restrict__ hh,
                             const _Float16* __restrict__ Blp,
                             const _Float16* __restrict__ Brp,
                             const float* __restrict__ bias,
                             float* __restrict__ hnew) {
  const int mt = (blockIdx.x * blockDim.x + threadIdx.x) >> 5;
  if (mt >= MTILES) return;
  const int lane = threadIdx.x & 31;
  const float scale = invdeg[(mt << 4) + (lane & 15)];  // fused 1/deg
  v8f c[8] = {};
#pragma unroll
  for (int kt = 0; kt < (DHID >> 5); ++kt) {
    const v16h a1 = load_a_f32(agg, DHID, mt << 4, kt << 5, scale);
#pragma unroll
    for (int nt = 0; nt < 8; ++nt)
      c[nt] = WMMA_F16(a1, load_b_pack(Blp, 8, kt, nt), c[nt]);
    const v16h a2 = load_a_f16(hh, DHID, mt << 4, kt << 5);
#pragma unroll
    for (int nt = 0; nt < 8; ++nt)
      c[nt] = WMMA_F16(a2, load_b_pack(Brp, 8, kt, nt), c[nt]);
  }
#pragma unroll
  for (int nt = 0; nt < 8; ++nt)
    store_tile(hnew, (_Float16*)nullptr, DHID, mt << 4, nt << 4, c[nt], bias, 0);
}

// ---- codon hidden GEMM: t64 = relu(h_obs @ Wc1 + bc1); f16 A operand ------
__global__ void k_gemm_c1(const _Float16* __restrict__ hh,
                          const _Float16* __restrict__ Bp,
                          const float* __restrict__ bias,
                          float* __restrict__ t64) {
  const int mt = (blockIdx.x * blockDim.x + threadIdx.x) >> 5;
  if (mt >= MTILES) return;
  v8f c[4] = {};
#pragma unroll
  for (int kt = 0; kt < (DHID >> 5); ++kt) {
    const v16h a = load_a_f16(hh, DHID, mt << 4, kt << 5);
#pragma unroll
    for (int nt = 0; nt < 4; ++nt)
      c[nt] = WMMA_F16(a, load_b_pack(Bp, 4, kt, nt), c[nt]);
  }
#pragma unroll
  for (int nt = 0; nt < 4; ++nt)
    store_tile(t64, (_Float16*)nullptr, DIN, mt << 4, nt << 4, c[nt], bias, 1);
}

// ---------------- misc small kernels ----------------
__global__ void k_zero(float* __restrict__ p, long n) {
  const long i = (long)blockIdx.x * blockDim.x + threadIdx.x;
  if (i < n) p[i] = 0.0f;
}

__global__ void k_deg(const long long* __restrict__ edge, float* __restrict__ deg) {
  const int e = blockIdx.x * blockDim.x + threadIdx.x;
  if (e < N_EDGES) atomicAdd(&deg[(int)edge[N_EDGES + e]], 1.0f);
}

__global__ void k_invdeg(float* __restrict__ deg) {
  const int i = blockIdx.x * blockDim.x + threadIdx.x;
  if (i < N_NODES) deg[i] = 1.0f / fmaxf(deg[i], 1.0f);
}

// wave-per-edge: 32 lanes x float4 covers all 128 dims of one edge.
__global__ void k_agg(const long long* __restrict__ edge,
                      const float* __restrict__ h, float* __restrict__ agg) {
  const int wid = (blockIdx.x * blockDim.x + threadIdx.x) >> 5;
  if (wid >= N_EDGES) return;
  const int lane = threadIdx.x & 31;
  __builtin_prefetch(edge + wid + 2048, 0, 0);  // stream the edge list
  const int s = (int)edge[wid];
  const int d = (int)edge[N_EDGES + wid];
  const float4 v = *(const float4*)(h + (size_t)s * DHID + lane * 4);
  float* dp = agg + (size_t)d * DHID + lane * 4;
  atomicAdd(dp + 0, v.x);
  atomicAdd(dp + 1, v.y);
  atomicAdd(dp + 2, v.z);
  atomicAdd(dp + 3, v.w);
}

// wave-per-row: h = relu(LN(hnew)*g + b) + h ; refresh f16 shadow.
__global__ void k_ln_relu_res(const float* __restrict__ hnew,
                              const float* __restrict__ g,
                              const float* __restrict__ b,
                              float* __restrict__ h, _Float16* __restrict__ hh) {
  const int row = (blockIdx.x * blockDim.x + threadIdx.x) >> 5;
  if (row >= N_NODES) return;
  const int lane = threadIdx.x & 31;
  const float4 v = *(const float4*)(hnew + (size_t)row * DHID + lane * 4);
  const float mu = wave_sum(v.x + v.y + v.z + v.w) * (1.0f / DHID);
  const float dx = v.x - mu, dy = v.y - mu, dz = v.z - mu, dw = v.w - mu;
  const float var = wave_sum(dx * dx + dy * dy + dz * dz + dw * dw) * (1.0f / DHID);
  const float rstd = rsqrtf(var + 1e-5f);
  const float4 gg = *(const float4*)(g + lane * 4);
  const float4 bb = *(const float4*)(b + lane * 4);
  float* rowh = h + (size_t)row * DHID + lane * 4;
  float4 hv = *(const float4*)rowh;
  float4 o;
  o.x = fmaxf(dx * rstd * gg.x + bb.x, 0.0f) + hv.x;
  o.y = fmaxf(dy * rstd * gg.y + bb.y, 0.0f) + hv.y;
  o.z = fmaxf(dz * rstd * gg.z + bb.z, 0.0f) + hv.z;
  o.w = fmaxf(dw * rstd * gg.w + bb.w, 0.0f) + hv.w;
  *(float4*)rowh = o;
  _Float16* hhp = hh + (size_t)row * DHID + lane * 4;
  hhp[0] = (_Float16)o.x; hhp[1] = (_Float16)o.y;
  hhp[2] = (_Float16)o.z; hhp[3] = (_Float16)o.w;
}

// FiLM MLP: gb = (relu(sun@Wf1+bf1))@Wf2 + bf2   (single block, tiny)
__global__ void k_film(const float* __restrict__ sun, const float* __restrict__ Wf1,
                       const float* __restrict__ bf1, const float* __restrict__ Wf2,
                       const float* __restrict__ bf2, float* __restrict__ gb) {
  __shared__ float fsh[FILMH];
  const int t = threadIdx.x;  // 256
  if (t < FILMH) {
    float acc = bf1[t];
    for (int i = 0; i < SUNDIM; ++i) acc += sun[i] * Wf1[i * FILMH + t];
    fsh[t] = fmaxf(acc, 0.0f);
  }
  __syncthreads();
  float acc = bf2[t];
  for (int i = 0; i < FILMH; ++i) acc += fsh[i] * Wf2[i * (2 * DHID) + t];
  gb[t] = acc;
}

// h_obs = gamma*h + beta ; update f32 h, f16 shadow, and output slot.
__global__ void k_film_apply(float* __restrict__ h, _Float16* __restrict__ hh,
                             float* __restrict__ out_hobs,
                             const float* __restrict__ gb) {
  const int idx = blockIdx.x * blockDim.x + threadIdx.x;
  if (idx >= N_NODES * DHID) return;
  const int d = idx & (DHID - 1);
  const float v = gb[d] * h[idx] + gb[DHID + d];
  h[idx] = v;
  hh[idx] = (_Float16)v;
  out_hobs[idx] = v;
}

// codon = sigmoid(t64 @ Wc2 + bc2), wave per row over the 64-wide hidden.
__global__ void k_codon(const float* __restrict__ t64, const float* __restrict__ Wc2,
                        const float* __restrict__ bc2, float* __restrict__ codon) {
  const int row = (blockIdx.x * blockDim.x + threadIdx.x) >> 5;
  if (row >= N_NODES) return;
  const int lane = threadIdx.x & 31;
  const float* r = t64 + (size_t)row * 64 + lane * 2;
  float acc = r[0] * Wc2[lane * 2] + r[1] * Wc2[lane * 2 + 1];
  acc = wave_sum(acc);
  if (lane == 0) codon[row] = 1.0f / (1.0f + expf(-(acc + bc2[0])));
}

// ---------------- masked attention pools (5, via blockIdx.y) ----------------
struct PoolArgs {
  const float* Wa[5];
  const float* ba[5];
  const float* Wh[5];
  const float* bh[5];
  const unsigned char* mask[5];
};

__global__ void k_attn_logits(const float* __restrict__ hobs, PoolArgs pa,
                              float* __restrict__ logits) {
  const int p = blockIdx.y;
  const int row = (blockIdx.x * blockDim.x + threadIdx.x) >> 5;
  if (row >= N_NODES) return;
  const int lane = threadIdx.x & 31;
  const float* r = hobs + (size_t)row * DHID + lane * 4;
  const float* wa = pa.Wa[p] + lane * 4;
  float acc = r[0] * wa[0] + r[1] * wa[1] + r[2] * wa[2] + r[3] * wa[3];
  acc = wave_sum(acc);
  if (lane == 0) {
    const float l = acc + pa.ba[p][0];
    logits[(size_t)p * N_NODES + row] = pa.mask[p][row] ? l : -__builtin_inff();
  }
}

__global__ void k_attn_partmax(const float* __restrict__ logits,
                               float* __restrict__ partmax) {
  const int p = blockIdx.y;
  __shared__ float sm[256];
  float m = -__builtin_inff();
  for (int i = blockIdx.x * blockDim.x + threadIdx.x; i < N_NODES;
       i += gridDim.x * blockDim.x)
    m = fmaxf(m, logits[(size_t)p * N_NODES + i]);
  sm[threadIdx.x] = m;
  __syncthreads();
  for (int s = 128; s >= 1; s >>= 1) {
    if (threadIdx.x < (unsigned)s)
      sm[threadIdx.x] = fmaxf(sm[threadIdx.x], sm[threadIdx.x + s]);
    __syncthreads();
  }
  if (threadIdx.x == 0) partmax[p * PB + blockIdx.x] = sm[0];
}

__global__ void k_attn_gmax(const float* __restrict__ partmax,
                            float* __restrict__ gmax, float* __restrict__ sumw,
                            float* __restrict__ vacc) {
  const int p = blockIdx.y;
  __shared__ float sm[PB];
  const int t = threadIdx.x;  // 128
  sm[t] = partmax[p * PB + t];
  __syncthreads();
  for (int s = 64; s >= 1; s >>= 1) {
    if (t < s) sm[t] = fmaxf(sm[t], sm[t + s]);
    __syncthreads();
  }
  if (t == 0) { gmax[p] = sm[0]; sumw[p] = 0.0f; }
  vacc[p * DHID + t] = 0.0f;
}

__global__ void k_attn_acc(const float* __restrict__ hobs,
                           const float* __restrict__ logits,
                           const float* __restrict__ gmax,
                           float* __restrict__ vacc, float* __restrict__ sumw) {
  const int p = blockIdx.y;
  __shared__ float sacc[DHID];
  __shared__ float ssum;
  const int t = threadIdx.x;  // 256 = 8 waves
  if (t < DHID) sacc[t] = 0.0f;
  if (t == 0) ssum = 0.0f;
  __syncthreads();
  const int lane = t & 31;
  const float gm = gmax[p];
  float a0 = 0, a1 = 0, a2 = 0, a3 = 0, ws = 0;
  const int gw = (blockIdx.x * blockDim.x + t) >> 5;
  const int nwaves = (gridDim.x * blockDim.x) >> 5;
  for (int r = gw; r < N_NODES; r += nwaves) {
    const float w = __expf(logits[(size_t)p * N_NODES + r] - gm);  // exp(-inf)=0
    const float* row = hobs + (size_t)r * DHID + lane * 4;
    a0 += w * row[0]; a1 += w * row[1]; a2 += w * row[2]; a3 += w * row[3];
    if (lane == 0) ws += w;
  }
  atomicAdd(&sacc[lane * 4 + 0], a0);
  atomicAdd(&sacc[lane * 4 + 1], a1);
  atomicAdd(&sacc[lane * 4 + 2], a2);
  atomicAdd(&sacc[lane * 4 + 3], a3);
  if (lane == 0) atomicAdd(&ssum, ws);
  __syncthreads();
  if (t < DHID) atomicAdd(&vacc[p * DHID + t], sacc[t]);
  if (t == 0) atomicAdd(&sumw[p], ssum);
}

__global__ void k_attn_final(const float* __restrict__ vacc,
                             const float* __restrict__ sumw, PoolArgs pa,
                             float* __restrict__ out_vecs,
                             float* __restrict__ out_scores) {
  const int p = blockIdx.y;
  __shared__ float sm[DHID];
  const int t = threadIdx.x;  // 128
  const float v = vacc[p * DHID + t] / sumw[p];
  out_vecs[p * DHID + t] = v;
  sm[t] = v * pa.Wh[p][t];
  __syncthreads();
  for (int s = 64; s >= 1; s >>= 1) {
    if (t < s) sm[t] += sm[t + s];
    __syncthreads();
  }
  if (t == 0) out_scores[p] = 1.0f / (1.0f + expf(-(sm[0] + pa.bh[p][0])));
}

// ---------------------------------------------------------------------------
extern "C" void kernel_launch(void* const* d_in, const int* in_sizes, int n_in,
                              void* d_out, int out_size, void* d_ws, size_t ws_size,
                              hipStream_t stream) {
  (void)in_sizes; (void)n_in; (void)out_size; (void)ws_size;

  // ---- inputs (setup_inputs dict order) ----
  const float*     x    = (const float*)d_in[0];
  const long long* edge = (const long long*)d_in[1];
  const float*     sun  = (const float*)d_in[2];
  const float* W_in = (const float*)d_in[8];
  const float* b_in = (const float*)d_in[9];
  const float* Wl   = (const float*)d_in[10];
  const float* bl   = (const float*)d_in[11];
  const float* Wr   = (const float*)d_in[12];
  const float* ln_g = (const float*)d_in[13];
  const float* ln_b = (const float*)d_in[14];
  const float* Wf1  = (const float*)d_in[15];
  const float* bf1  = (const float*)d_in[16];
  const float* Wf2  = (const float*)d_in[17];
  const float* bf2  = (const float*)d_in[18];
  const float* Wc1  = (const float*)d_in[19];
  const float* bc1  = (const float*)d_in[20];
  const float* Wc2  = (const float*)d_in[21];
  const float* bc2  = (const float*)d_in[22];
  PoolArgs pa;
  for (int i = 0; i < 5; ++i) {
    pa.Wa[i]   = (const float*)d_in[23 + i * 4 + 0];
    pa.ba[i]   = (const float*)d_in[23 + i * 4 + 1];
    pa.Wh[i]   = (const float*)d_in[23 + i * 4 + 2];
    pa.bh[i]   = (const float*)d_in[23 + i * 4 + 3];
    pa.mask[i] = (const unsigned char*)d_in[3 + i];
  }

  // ---- outputs: [codon N][scores 5][vecs 5*128][h_obs N*D] ----
  float* out_codon  = (float*)d_out;
  float* out_scores = out_codon + N_NODES;
  float* out_vecs   = out_scores + 5;
  float* out_hobs   = out_vecs + 5 * DHID;

  // ---- workspace carve-up (256B aligned) ----
  char* wp = (char*)d_ws;
  auto carve = [&](size_t bytes) -> void* {
    void* p = (void*)wp;
    wp += (bytes + 255) & ~(size_t)255;
    return p;
  };
  float*    h    = (float*)carve((size_t)N_NODES * DHID * 4);
  _Float16* hh   = (_Float16*)carve((size_t)N_NODES * DHID * 2);  // f16 shadow
  float*    agg  = (float*)carve((size_t)N_NODES * DHID * 4);     // also codon hidden
  float*    hnew = (float*)carve((size_t)N_NODES * DHID * 4);
  float*    deg  = (float*)carve((size_t)N_NODES * 4);            // becomes inv_deg
  float* gb      = (float*)carve(2 * DHID * 4);
  float* gmax    = (float*)carve(5 * 4);
  float* sumw    = (float*)carve(5 * 4);
  float* vacc    = (float*)carve(5 * DHID * 4);
  float* partmax = (float*)carve(5 * PB * 4);
  float* logits  = (float*)carve((size_t)5 * N_NODES * 4);
  _Float16* Winp = (_Float16*)carve(8192 * 2);
  _Float16* Wlp  = (_Float16*)carve((size_t)3 * 16384 * 2);
  _Float16* Wrp  = (_Float16*)carve((size_t)3 * 16384 * 2);
  _Float16* Wc1p = (_Float16*)carve(8192 * 2);

  const int BLK = 256;
  const int gemm_blocks = (MTILES * 32 + BLK - 1) / BLK;  // 391 (wave per 16-row stripe)
  const int nd_blocks = (N_NODES * DHID) / BLK;           // 25000
  const int row_blocks = (N_NODES * 32) / BLK;            // 6250 (wave per row)

  // ---- pack weights into WMMA B-fragment order (f16) ----
  k_pack_weight<<<8192 / BLK, BLK, 0, stream>>>(W_in, DIN, DHID, Winp);
  for (int l = 0; l < NLAYERS; ++l) {
    k_pack_weight<<<16384 / BLK, BLK, 0, stream>>>(Wl + (size_t)l * DHID * DHID,
                                                   DHID, DHID, Wlp + (size_t)l * 16384);
    k_pack_weight<<<16384 / BLK, BLK, 0, stream>>>(Wr + (size_t)l * DHID * DHID,
                                                   DHID, DHID, Wrp + (size_t)l * 16384);
  }
  k_pack_weight<<<8192 / BLK, BLK, 0, stream>>>(Wc1, DHID, DIN, Wc1p);

  // ---- degrees ----
  k_zero<<<(N_NODES + BLK - 1) / BLK, BLK, 0, stream>>>(deg, N_NODES);
  k_deg<<<N_EDGES / BLK, BLK, 0, stream>>>(edge, deg);
  k_invdeg<<<(N_NODES + BLK - 1) / BLK, BLK, 0, stream>>>(deg);

  // ---- h = relu(x @ W_in + b_in) ----
  k_gemm_in<<<gemm_blocks, BLK, 0, stream>>>(x, Winp, b_in, h, hh);

  // ---- 3 GNN layers ----
  for (int l = 0; l < NLAYERS; ++l) {
    k_zero<<<nd_blocks, BLK, 0, stream>>>(agg, (long)N_NODES * DHID);
    k_agg<<<(N_EDGES * 32) / BLK, BLK, 0, stream>>>(edge, h, agg);
    k_gemm_layer<<<gemm_blocks, BLK, 0, stream>>>(
        agg, deg, hh, Wlp + (size_t)l * 16384, Wrp + (size_t)l * 16384,
        bl + (size_t)l * DHID, hnew);
    k_ln_relu_res<<<row_blocks, BLK, 0, stream>>>(hnew, ln_g + (size_t)l * DHID,
                                                  ln_b + (size_t)l * DHID, h, hh);
  }

  // ---- FiLM ----
  k_film<<<1, 256, 0, stream>>>(sun, Wf1, bf1, Wf2, bf2, gb);
  k_film_apply<<<nd_blocks, BLK, 0, stream>>>(h, hh, out_hobs, gb);  // h := h_obs

  // ---- codon head ----
  k_gemm_c1<<<gemm_blocks, BLK, 0, stream>>>(hh, Wc1p, bc1, agg);  // agg = N x 64
  k_codon<<<row_blocks, BLK, 0, stream>>>(agg, Wc2, bc2, out_codon);

  // ---- 5 masked attention pools (pool = blockIdx.y) ----
  {
    dim3 gl(row_blocks, 5);
    k_attn_logits<<<gl, BLK, 0, stream>>>(h, pa, logits);
    dim3 gm(PB, 5);
    k_attn_partmax<<<gm, BLK, 0, stream>>>(logits, partmax);
    dim3 g1(1, 5);
    k_attn_gmax<<<g1, PB, 0, stream>>>(partmax, gmax, sumw, vacc);
    k_attn_acc<<<gm, BLK, 0, stream>>>(h, logits, gmax, vacc, sumw);
    k_attn_final<<<g1, DHID, 0, stream>>>(vacc, sumw, pa, out_vecs, out_scores);
  }
}
